// OTHead_35673998360861
// MI455X (gfx1250) — matrix operand order repaired and checked by reading, not compiled, not run on hardware
//
#include <hip/hip_runtime.h>

typedef _Float16 f16;
typedef __attribute__((ext_vector_type(16))) _Float16 v16h;
typedef __attribute__((ext_vector_type(8)))  float    v8f;
typedef __attribute__((ext_vector_type(2)))  float    v2f;

#define N_IMG   64
#define N_PATCH 48
#define DV      2048
#define N_TXT   64
#define N_WORD  64
#define DT      768
#define E_DIM   256
#define IPOT_ITER 50
#define BIG_MASK  1e4f

// ---------------------------------------------------------------------------
// Kernel 1: Y[16 rows x 256] = X[16 x K] @ W[K x 256] + b, L2-normalize rows,
// store f16. Full f32 WMMA (V_WMMA_F32_16X16X4_F32) so it is numerically the
// reference's f32 GEMM. 8 waves/block; wave w owns column tiles {w, w+8}.
// X chunks are async-staged to LDS (GLOBAL_LOAD_ASYNC_TO_LDS_B128, ASYNCcnt);
// W (L2-resident) streams straight into B fragments with coalesced loads.
// ---------------------------------------------------------------------------
__global__ __launch_bounds__(256)
void embed_norm_wmma_kernel(const float* __restrict__ X,
                            const float* __restrict__ W,
                            const float* __restrict__ bias,
                            f16* __restrict__ out, int K) {
  __shared__ __align__(16) float Xl[16 * 256];   // 16 KB: K-chunk / out tile
  __shared__ float nrm[16];
  const int tid  = threadIdx.x;
  const int wid  = tid >> 5;
  const int lane = tid & 31;
  const int l16  = lane & 15;
  const int h16  = lane >> 4;
  const int r0   = blockIdx.x * 16;              // first output row

  v8f c0 = {0.f,0.f,0.f,0.f,0.f,0.f,0.f,0.f};
  v8f c1 = {0.f,0.f,0.f,0.f,0.f,0.f,0.f,0.f};

  const unsigned long long xbase = (unsigned long long)(X + (size_t)r0 * K);
  const unsigned int lds0 = (unsigned int)(unsigned long long)(&Xl[0]);

  const int nchunk = K >> 8;                     // K / 256
  for (int ch = 0; ch < nchunk; ++ch) {
    // ---- async-stage X[16 rows][256 k] chunk straight into LDS ----
#pragma unroll
    for (int q = 0; q < 4; ++q) {
      const int idx = tid + q * 256;             // 0..1023 (float4 units)
      const int row = idx >> 6;                  // 64 float4 per row
      const int col = idx & 63;
      const unsigned int goff = (unsigned int)((row * K + ch * 256 + col * 4) * 4);
      const unsigned int loff = (unsigned int)(idx * 16);
      asm volatile("global_load_async_to_lds_b128 %0, %1, %2"
                   :: "v"(lds0 + loff), "v"(goff), "s"(xbase) : "memory");
    }
    asm volatile("s_wait_asynccnt 0x0" ::: "memory");
    __syncthreads();

    // ---- 64 steps of K=4 f32 WMMA ----
    // A 16x4 f32 layout: lane l16 = row M; lanes 0-15 K={0,1}, 16-31 K={2,3}
    const float* arow = Xl + l16 * 256 + h16 * 2;
    const float* wb   = W + ((size_t)(ch * 256 + h16 * 2)) * E_DIM;
#pragma unroll 8
    for (int kk = 0; kk < 64; ++kk) {
      v2f a, b0, b1;
      a.x = arow[kk * 4 + 0];
      a.y = arow[kk * 4 + 1];
      const float* wrow = wb + (size_t)(kk * 4) * E_DIM;
      b0.x = wrow[wid * 16 + l16];
      b0.y = wrow[E_DIM + wid * 16 + l16];
      b1.x = wrow[(wid + 8) * 16 + l16];
      b1.y = wrow[E_DIM + (wid + 8) * 16 + l16];
      c0 = __builtin_amdgcn_wmma_f32_16x16x4_f32(false, a, false, b0,
                                                 (short)0, c0, false, false);
      c1 = __builtin_amdgcn_wmma_f32_16x16x4_f32(false, a, false, b1,
                                                 (short)0, c1, false, false);
    }
    __syncthreads();   // chunk reads done before restaging LDS
  }

  // ---- spill f32 tile (+bias) to LDS for the row-norm pass ----
  {
    const float be0 = bias[wid * 16 + l16];
    const float be1 = bias[(wid + 8) * 16 + l16];
#pragma unroll
    for (int r = 0; r < 8; ++r) {
      const int m = h16 * 8 + r;                 // D layout: lanes16-31 -> M+8
      Xl[m * 256 + wid * 16 + l16]       = c0[r] + be0;
      Xl[m * 256 + (wid + 8) * 16 + l16] = c1[r] + be1;
    }
  }
  __syncthreads();

  // ---- row L2 norms: 16 lanes per row, shuffle reduce ----
  {
    const int m = tid >> 4, j = tid & 15;
    float s = 0.f;
    for (int e = j; e < 256; e += 16) { const float v = Xl[m * 256 + e]; s = fmaf(v, v, s); }
    s += __shfl_xor(s, 1, 32);
    s += __shfl_xor(s, 2, 32);
    s += __shfl_xor(s, 4, 32);
    s += __shfl_xor(s, 8, 32);
    if (j == 0) nrm[m] = 1.0f / fmaxf(sqrtf(s), 1e-5f);
  }
  __syncthreads();

  // ---- normalize, cast f16, coalesced store ----
  for (int q = 0; q < 16; ++q) {
    const int idx = tid + q * 256;               // 0..4095
    const int m = idx >> 8, e = idx & 255;
    out[(size_t)(r0 + m) * E_DIM + e] = (f16)(Xl[idx] * nrm[m]);
  }
}

// ---------------------------------------------------------------------------
// Kernel 2: one workgroup per (text t, image i) pair.
// Phase 0: async global->LDS staging (GLOBAL_LOAD_ASYNC_TO_LDS_B128, ASYNCcnt)
// Phase 1: WMMA cosine cost  C[m][p] = 1 - <wn[t,m], pn[i,p]>  (K = 256)
// Phase 2: 50 IPOT iterations entirely in LDS, then trace(C @ T).
// ---------------------------------------------------------------------------
__global__ __launch_bounds__(512)
void ot_kernel(const f16* __restrict__ wn_g, const f16* __restrict__ pn_g,
               const int* __restrict__ txt_lens, float* __restrict__ out) {
  __shared__ __align__(16) char smem[57344];
  // phase-1 view: f16 embeddings
  f16* wn_lds = (f16*)smem;                 // 64 x 256 halves = 32768 B
  f16* pn_lds = (f16*)(smem + 32768);       // 48 x 256 halves = 24576 B
  // phase-2 view (aliased; all phase-1 reads complete before reuse)
  float* Cm    = (float*)smem;              // [64][48] cost          12288 B
  float* Am    = (float*)(smem + 12288);    // [48][64] exp kernel    12288 B
  float* Tm    = (float*)(smem + 24576);    // [48][64] plan          12288 B
  float* Qm    = (float*)(smem + 36864);    // [48][64] A*T           12288 B
  float* sigma = (float*)(smem + 49152);    // [64]
  float* delta = (float*)(smem + 49408);    // [48]
  float* wred  = (float*)(smem + 49600);    // [16]

  const int i   = blockIdx.x;   // image
  const int t   = blockIdx.y;   // text
  const int tid = threadIdx.x;
  const int len = txt_lens[t];

  // ---- Phase 0: direct async global->LDS staging (no VGPR round-trip) ----
  {
    const unsigned long long wbase =
        (unsigned long long)(wn_g + (size_t)t * N_WORD * E_DIM);
    const unsigned long long pbase =
        (unsigned long long)(pn_g + (size_t)i * N_PATCH * E_DIM);
    const unsigned int lds0 = (unsigned int)(unsigned long long)(&smem[0]);
#pragma unroll
    for (int k = 0; k < 4; ++k) {              // wn: 2048 x b128
      unsigned int off = (unsigned int)((tid + k * 512) * 16);
      asm volatile("global_load_async_to_lds_b128 %0, %1, %2"
                   :: "v"(lds0 + off), "v"(off), "s"(wbase) : "memory");
    }
#pragma unroll
    for (int k = 0; k < 3; ++k) {              // pn: 1536 x b128
      unsigned int off = (unsigned int)((tid + k * 512) * 16);
      asm volatile("global_load_async_to_lds_b128 %0, %1, %2"
                   :: "v"(lds0 + 32768u + off), "v"(off), "s"(pbase) : "memory");
    }
    asm volatile("s_wait_asynccnt 0x0" ::: "memory");
  }
  __syncthreads();

  // ---- Phase 1: WMMA, 4x3 tiles of 16x16, waves 0..11 own one tile each ----
  const int wid  = tid >> 5;
  const int lane = tid & 31;
  const int l16  = lane & 15;
  const int h16  = lane >> 4;
  v8f c = {0.f,0.f,0.f,0.f,0.f,0.f,0.f,0.f};
  int mt = 0, pt = 0;
  if (wid < 12) {
    mt = wid / 3;              // word-tile   (M dim, 4 tiles)
    pt = wid - mt * 3;         // patch-tile  (N dim, 3 tiles)
    // A: lanes 0-15 hold K=0..15, lanes 16-31 hold K=16..31
    const f16* wrow = wn_lds + (mt * 16 + l16) * E_DIM + h16 * 16;
    // B: col p = pt*16+l16; lanes 0-15 K=0..15, lanes 16-31 K=16..31
    const f16* pcol = pn_lds + (pt * 16 + l16) * E_DIM + h16 * 16;
#pragma unroll
    for (int kb = 0; kb < 8; ++kb) {
      union { v16h h; uint4 u[2]; } Af, Bf;
      const uint4* wa = (const uint4*)(wrow + kb * 32);
      Af.u[0] = wa[0];          // K = kbase + 0..7
      Af.u[1] = wa[1];          // K = kbase + 8..15
      const uint4* pb = (const uint4*)(pcol + kb * 32);
      Bf.u[0] = pb[0];          // K = kbase + 0..7
      Bf.u[1] = pb[1];          // K = kbase + 8..15
      c = __builtin_amdgcn_wmma_f32_16x16x32_f16(
              false, Af.h, false, Bf.h, (short)0, c, false, false);
    }
  }
  __syncthreads();   // all LDS reads of embeddings done -> safe to alias

  // ---- masked cost into LDS: C[m][p] = pad(m) ? 0 : 1 - cos ----
  if (wid < 12) {
#pragma unroll
    for (int r = 0; r < 8; ++r) {
      const int m = mt * 16 + h16 * 8 + r;   // D layout: lanes16-31 -> M+8
      const int p = pt * 16 + l16;
      Cm[m * N_PATCH + p] = (m < len) ? (1.0f - c[r]) : 0.0f;
    }
  }
  __syncthreads();

  // ---- IPOT init: A = exp(-C^T/beta) masked, T0 = !pad, sigma0 = 1/len ----
  for (int idx = tid; idx < N_PATCH * N_WORD; idx += 512) {
    const int n = idx >> 6, m = idx & 63;
    const bool pad = (m >= len);
    Am[idx] = pad ? 0.f : __expf(-2.0f * Cm[m * N_PATCH + n]);  // beta = 0.5
    Tm[idx] = pad ? 0.f : 1.f;
  }
  if (tid < N_WORD) sigma[tid] = (tid < len) ? (1.0f / (float)len) : 0.f;
  __syncthreads();

  const float xlen = (float)len;
  for (int it = 0; it < IPOT_ITER; ++it) {
    for (int idx = tid; idx < N_PATCH * N_WORD; idx += 512)
      Qm[idx] = Am[idx] * Tm[idx];
    __syncthreads();
    // delta[n] = 1 / (48 * sum_m Q[n,m]*sigma[m])   (48 rows x 8 lanes)
    if (tid < N_PATCH * 8) {
      const int n = tid >> 3, j = tid & 7;
      const float* qn = Qm + n * 64;
      float s = 0.f;
      for (int m = j; m < 64; m += 8) s = fmaf(qn[m], sigma[m], s);
      s += __shfl_xor(s, 1, 32);
      s += __shfl_xor(s, 2, 32);
      s += __shfl_xor(s, 4, 32);
      if (j == 0) delta[n] = 1.0f / (48.0f * s);
    }
    __syncthreads();
    // sigma[m] = 1 / (len * sum_n delta[n]*Q[n,m] + mask[m])  (64 x 8 lanes)
    {
      const int m = tid >> 3, j = tid & 7;
      float s = 0.f;
      for (int n = j; n < N_PATCH; n += 8) s = fmaf(Qm[n * 64 + m], delta[n], s);
      s += __shfl_xor(s, 1, 32);
      s += __shfl_xor(s, 2, 32);
      s += __shfl_xor(s, 4, 32);
      if (j == 0) {
        const float msk = (m >= len) ? BIG_MASK : 0.f;
        sigma[m] = 1.0f / (xlen * s + msk);
      }
    }
    __syncthreads();
    for (int idx = tid; idx < N_PATCH * N_WORD; idx += 512) {
      const int n = idx >> 6, m = idx & 63;
      Tm[idx] = delta[n] * Qm[idx] * sigma[m];
    }
    __syncthreads();
  }

  // ---- dist = trace(C @ T) = sum_{m,n} C[m,n]*T[n,m]; out = 1 - dist ----
  float part = 0.f;
  for (int idx = tid; idx < N_PATCH * N_WORD; idx += 512) {
    const int n = idx >> 6, m = idx & 63;
    part = fmaf(Cm[m * N_PATCH + n], Tm[idx], part);
  }
  for (int off = 16; off; off >>= 1) part += __shfl_xor(part, off, 32);
  if (lane == 0) wred[wid] = part;
  __syncthreads();
  if (tid == 0) {
    float d = 0.f;
    for (int w = 0; w < 16; ++w) d += wred[w];
    out[t * N_IMG + i] = 1.0f - d;
  }
}

// ---------------------------------------------------------------------------
extern "C" void kernel_launch(void* const* d_in, const int* in_sizes, int n_in,
                              void* d_out, int out_size, void* d_ws, size_t ws_size,
                              hipStream_t stream) {
  const float* patch_feat = (const float*)d_in[0];  // [64,48,2048]
  const float* word_feat  = (const float*)d_in[1];  // [64,64,768]
  const int*   txt_lens   = (const int*)d_in[2];    // [64]
  const float* Wv = (const float*)d_in[3];          // [2048,256]
  const float* bv = (const float*)d_in[4];          // [256]
  const float* Wt = (const float*)d_in[5];          // [768,256]
  const float* bt = (const float*)d_in[6];          // [256]
  float* out = (float*)d_out;                       // [64,64]

  // workspace: f16 normalized embeddings (L2-resident)
  f16* pn = (f16*)d_ws;                                        // 3072*256
  f16* wn = (f16*)((char*)d_ws + (size_t)3072 * 256 * 2);      // 4096*256

  embed_norm_wmma_kernel<<<(N_IMG * N_PATCH) / 16, 256, 0, stream>>>(
      patch_feat, Wv, bv, pn, DV);
  embed_norm_wmma_kernel<<<(N_TXT * N_WORD) / 16, 256, 0, stream>>>(
      word_feat, Wt, bt, wn, DT);
  ot_kernel<<<dim3(N_IMG, N_TXT), 512, 0, stream>>>(wn, pn, txt_lens, out);
}
